// ColBERTReranker_45045617000829
// MI455X (gfx1250) — matrix operand order repaired and checked by reading, not compiled, not run on hardware
//
#include <hip/hip_runtime.h>

// ---------------------------------------------------------------------------
// ColBERT MaxSim reranker for gfx1250 (MI455X, wave32, WMMA).
//   scores[q][n] = sum_s max_t <q[q,s,:], d[n,t,:]>
// Q=16, S=32, N=1024, T=256, D=128.  One workgroup (8 waves) per doc n.
// bf16 WMMA (16x16x32) with f32 accumulate; doc tile staged via LDS in two
// 128-row chunks.  M-tiles processed in pairs so each B-fragment feeds 2 WMMAs.
// nt-unroll capped at 2 + launch bounds to stay under 256 VGPRs (no vgpr_msb).
// ---------------------------------------------------------------------------

typedef __attribute__((ext_vector_type(16))) __bf16 v16bf;
typedef __attribute__((ext_vector_type(8)))  float  v8f;

struct B32x2 { uint4 lo; uint4 hi; };
static_assert(sizeof(B32x2) == 32, "B32x2 must be 32 bytes");
static_assert(sizeof(v16bf) == 32, "v16bf must be 32 bytes");

#define QN      16
#define DQ      32
#define NDOC    1024
#define DDOC    256
#define DK      128
#define LDS_ROW 136   // 128 bf16 payload + 8 bf16 pad (272B row, bank-skewed)

__device__ __forceinline__ float fmax_raw(float a, float b) {
  float d;
  asm("v_max_num_f32 %0, %1, %2" : "=v"(d) : "v"(a), "v"(b));
  return d;
}

// Pack two f32 into packed bf16x2 (lo = a, hi = b).
__device__ __forceinline__ unsigned pack2(float a, float b) {
#if __has_builtin(__builtin_amdgcn_cvt_pk_bf16_f32)
  typedef __attribute__((ext_vector_type(2))) __bf16 v2bf;
  v2bf p = __builtin_amdgcn_cvt_pk_bf16_f32(a, b);
  return __builtin_bit_cast(unsigned, p);
#else
  unsigned short ua = __builtin_bit_cast(unsigned short, (__bf16)a);
  unsigned short ub = __builtin_bit_cast(unsigned short, (__bf16)b);
  return (unsigned)ua | ((unsigned)ub << 16);
#endif
}

__device__ __forceinline__ uint4 pack8(float4 f0, float4 f1) {
  uint4 u;
  u.x = pack2(f0.x, f0.y);
  u.y = pack2(f0.z, f0.w);
  u.z = pack2(f1.x, f1.y);
  u.w = pack2(f1.z, f1.w);
  return u;
}

// A-fragment (16x32 bf16) for v_wmma_f32_16x16x32_bf16.
// lane l<16  : row l,  elements 0..15 = K kbase+0..15
// lane l+16  : row l,  elements 0..15 = K kbase+16..31
__device__ __forceinline__ v16bf load_a_frag(const float* qrow, int kbase, int half) {
  const float4* p0 = (const float4*)(qrow + kbase + half * 16);
  const float4* p1 = (const float4*)(qrow + kbase + half * 16 + 8);
  B32x2 s;
  s.lo = pack8(p0[0], p0[1]);
  s.hi = pack8(p1[0], p1[1]);
  return __builtin_bit_cast(v16bf, s);
}

// B-fragment (32x16 bf16): lane c<16 = column c, K ks*32+0..15;
// lane c+16 = column c, K ks*32+16..31.  16 contiguous bf16 per lane.
__device__ __forceinline__ v16bf load_b_frag(const __bf16* row, int ks, int half) {
  const uint4* p = (const uint4*)(row + ks * 32 + half * 16);
  B32x2 s;
  s.lo = p[0];
  s.hi = p[1];
  return __builtin_bit_cast(v16bf, s);
}

// Process a pair of adjacent M-tiles (tm0 even, tm0+1) against all 8 N-tiles
// of the current LDS doc chunk.  Both tiles belong to query q = tm0>>1
// (s-rows 0..15 and 16..31).  B fragments are loaded once and feed 8 WMMAs.
__device__ __forceinline__ void process_pair(const float* __restrict__ qemb,
                                             const __bf16 (*sD)[LDS_ROW],
                                             int tm0, int c, int half,
                                             v8f& rmax0, v8f& rmax1) {
  const int qi = tm0 >> 1;
  const float* qrow0 = qemb + (size_t)(qi * DQ +      c) * DK;  // s = 0..15
  const float* qrow1 = qemb + (size_t)(qi * DQ + 16 + c) * DK;  // s = 16..31

  const v16bf a0 = load_a_frag(qrow0,  0, half);
  const v16bf a1 = load_a_frag(qrow0, 32, half);
  const v16bf a2 = load_a_frag(qrow0, 64, half);
  const v16bf a3 = load_a_frag(qrow0, 96, half);
  const v16bf a4 = load_a_frag(qrow1,  0, half);
  const v16bf a5 = load_a_frag(qrow1, 32, half);
  const v16bf a6 = load_a_frag(qrow1, 64, half);
  const v16bf a7 = load_a_frag(qrow1, 96, half);

  // Unroll-by-2: ~16 ds_load_b128 in flight ahead of 16 WMMAs, keeps B
  // register footprint bounded (~64 VGPRs) so the wave stays under 256.
  #pragma unroll 2
  for (int nt = 0; nt < 8; ++nt) {
    const __bf16* brow = &sD[nt * 16 + c][0];
    const v16bf b0 = load_b_frag(brow, 0, half);
    const v16bf b1 = load_b_frag(brow, 1, half);
    const v16bf b2 = load_b_frag(brow, 2, half);
    const v16bf b3 = load_b_frag(brow, 3, half);

    v8f acc0 = {0.f, 0.f, 0.f, 0.f, 0.f, 0.f, 0.f, 0.f};
    v8f acc1 = {0.f, 0.f, 0.f, 0.f, 0.f, 0.f, 0.f, 0.f};
    acc0 = __builtin_amdgcn_wmma_f32_16x16x32_bf16(false, a0, false, b0, (short)0, acc0, false, false);
    acc1 = __builtin_amdgcn_wmma_f32_16x16x32_bf16(false, a4, false, b0, (short)0, acc1, false, false);
    acc0 = __builtin_amdgcn_wmma_f32_16x16x32_bf16(false, a1, false, b1, (short)0, acc0, false, false);
    acc1 = __builtin_amdgcn_wmma_f32_16x16x32_bf16(false, a5, false, b1, (short)0, acc1, false, false);
    acc0 = __builtin_amdgcn_wmma_f32_16x16x32_bf16(false, a2, false, b2, (short)0, acc0, false, false);
    acc1 = __builtin_amdgcn_wmma_f32_16x16x32_bf16(false, a6, false, b2, (short)0, acc1, false, false);
    acc0 = __builtin_amdgcn_wmma_f32_16x16x32_bf16(false, a3, false, b3, (short)0, acc0, false, false);
    acc1 = __builtin_amdgcn_wmma_f32_16x16x32_bf16(false, a7, false, b3, (short)0, acc1, false, false);

    #pragma unroll
    for (int r = 0; r < 8; ++r) {
      rmax0[r] = fmax_raw(rmax0[r], acc0[r]);
      rmax1[r] = fmax_raw(rmax1[r], acc1[r]);
    }
  }
}

// Row-wise max across the 16 columns held per lane group, then sum the 16 rows
// of the tile.  C/D layout: VGPR r -> row r (lanes 0-15) / row r+8 (lanes
// 16-31), column = lane%16.
__device__ __forceinline__ float reduce_tile(v8f m) {
  float ssum = 0.f;
  #pragma unroll
  for (int r = 0; r < 8; ++r) {
    float v = m[r];
    v = fmax_raw(v, __shfl_xor(v, 1, 32));
    v = fmax_raw(v, __shfl_xor(v, 2, 32));
    v = fmax_raw(v, __shfl_xor(v, 4, 32));
    v = fmax_raw(v, __shfl_xor(v, 8, 32));
    ssum += v;                    // lanes 0-15: rows 0..7; lanes 16-31: rows 8..15
  }
  ssum += __shfl_xor(ssum, 16, 32);
  return ssum;
}

__global__ __launch_bounds__(256, 4)   // >=4 waves/SIMD -> allocator stays <=256 VGPRs
void colbert_maxsim_kernel(const float* __restrict__ qemb,
                           const float* __restrict__ demb,
                           float* __restrict__ out) {
  __shared__ __align__(16) __bf16 sD[128][LDS_ROW];   // ~34.8 KB doc chunk (bf16)
  __shared__ float sPartial[32];

  const int n    = (int)blockIdx.x;     // doc index
  const int tid  = (int)threadIdx.x;
  const int lane = tid & 31;
  const int wave = tid >> 5;            // 0..7
  const int half = lane >> 4;           // 0 or 1 (16-lane group)
  const int c    = lane & 15;

  const float NEG = -__builtin_inff();
  v8f rmax0 = {NEG, NEG, NEG, NEG, NEG, NEG, NEG, NEG};
  v8f rmax1 = rmax0;
  v8f rmax2 = rmax0;
  v8f rmax3 = rmax0;

  #pragma unroll 1
  for (int chunk = 0; chunk < 2; ++chunk) {
    // ---- stage 128 doc rows (f32 -> bf16) into LDS: 2 threads per row ----
    {
      const int r = tid >> 1;
      const int h = tid & 1;
      const float4* src =
          (const float4*)(demb + ((size_t)n * DDOC + (size_t)chunk * 128 + r) * DK + h * 64);
      #pragma unroll
      for (int i = 0; i < 8; ++i) {
        float4 f0 = src[2 * i];
        float4 f1 = src[2 * i + 1];
        *(uint4*)&sD[r][h * 64 + i * 8] = pack8(f0, f1);
      }
    }
    __syncthreads();

    // Wave owns M-tiles wave*4 .. wave*4+3, processed as two statically-
    // unrolled pairs -> running max lives in named v8f regs (no movrel).
    process_pair(qemb, sD, wave * 4 + 0, c, half, rmax0, rmax1);
    process_pair(qemb, sD, wave * 4 + 2, c, half, rmax2, rmax3);

    __syncthreads();   // protect LDS before next chunk overwrite
  }

  float t0 = reduce_tile(rmax0);
  float t1 = reduce_tile(rmax1);
  float t2 = reduce_tile(rmax2);
  float t3 = reduce_tile(rmax3);
  if (lane == 0) {
    sPartial[wave * 4 + 0] = t0;
    sPartial[wave * 4 + 1] = t1;
    sPartial[wave * 4 + 2] = t2;
    sPartial[wave * 4 + 3] = t3;
  }
  __syncthreads();

  // Query q's 32 s-rows = M-tiles 2q and 2q+1; 16 threads emit scores[:, n]
  if (tid < QN) {
    out[(size_t)tid * NDOC + n] = sPartial[2 * tid] + sPartial[2 * tid + 1];
  }
}

// ---------------------------------------------------------------------------
extern "C" void kernel_launch(void* const* d_in, const int* in_sizes, int n_in,
                              void* d_out, int out_size, void* d_ws, size_t ws_size,
                              hipStream_t stream) {
  (void)in_sizes; (void)n_in; (void)out_size; (void)d_ws; (void)ws_size;
  const float* qemb = (const float*)d_in[0];   // (16, 32, 128) f32
  const float* demb = (const float*)d_in[1];   // (1024, 256, 128) f32
  float*       out  = (float*)d_out;           // (16, 1024) f32

  dim3 grid(NDOC);
  dim3 block(256);
  hipLaunchKernelGGL(colbert_maxsim_kernel, grid, block, 0, stream, qemb, demb, out);
}
